// CausalSelfAttention_23021024706844
// MI455X (gfx1250) — matrix operand order repaired and checked by reading, not compiled, not run on hardware
//
#include <hip/hip_runtime.h>
#include <hip/hip_bf16.h>
#include <cstdint>
#include <cstddef>

// ---------------------------------------------------------------------------
// CDNA5 (gfx1250) causal self-attention: bf16 WMMA GEMMs + flash attention,
// with double-buffered async global->LDS staging (ASYNCcnt threshold waits).
// ---------------------------------------------------------------------------

typedef __bf16 v16bf __attribute__((ext_vector_type(16)));
typedef float  v8f   __attribute__((ext_vector_type(8)));
typedef int    v4i_a __attribute__((ext_vector_type(4)));

union Frag16 {
    v16bf          bf;
    uint4          u4[2];
    unsigned short us[16];
};

#if defined(__has_builtin)
#  if __has_builtin(__builtin_amdgcn_global_load_async_to_lds_b128)
#    define USE_ASYNC_LDS 1
#  endif
#endif

#ifdef USE_ASYNC_LDS
// Signature (from hipcc diagnostics): (int4 AS1* src, int4 AS3* dst, Ii, Ii)
#define ASYNC_CP16(gptr, lptr)                                                \
    __builtin_amdgcn_global_load_async_to_lds_b128(                           \
        (__attribute__((address_space(1))) v4i_a*)(gptr),                     \
        (__attribute__((address_space(3))) v4i_a*)(lptr), 0, 0)
#if __has_builtin(__builtin_amdgcn_s_wait_asynccnt)
#define ASYNC_WAIT(n) __builtin_amdgcn_s_wait_asynccnt(n)
#else
#define ASYNC_WAIT(n) asm volatile("s_wait_asynccnt %0" :: "i"(n) : "memory")
#endif
#endif

__device__ __forceinline__ unsigned short f32_bf16(float f) {
    union { float f; unsigned u; } x; x.f = f;
    unsigned u = x.u;
    u += 0x7fffu + ((u >> 16) & 1u);   // round-to-nearest-even
    return (unsigned short)(u >> 16);
}

// Exchange value with the other 16-lane half (lane ^ 16), wave32.
__device__ __forceinline__ float xhalf(float v) {
#if defined(__has_builtin) && __has_builtin(__builtin_amdgcn_permlanex16)
    unsigned u = __builtin_bit_cast(unsigned, v);
    u = __builtin_amdgcn_permlanex16(u, u, 0x76543210u, 0xFEDCBA98u, false, false);
    return __builtin_bit_cast(float, u);
#else
    return __shfl_xor(v, 16, 32);
#endif
}

// ---------------------------------------------------------------------------
// fp32 -> bf16 cast, 4 elements/thread (all sizes are multiples of 4)
// ---------------------------------------------------------------------------
__global__ void cast_f32_to_bf16_v4(const float* __restrict__ s,
                                    unsigned short* __restrict__ d, int n4) {
    int i = blockIdx.x * blockDim.x + threadIdx.x;
    if (i >= n4) return;
    float4 v = ((const float4*)s)[i];
    ushort4 o;
    o.x = f32_bf16(v.x); o.y = f32_bf16(v.y);
    o.z = f32_bf16(v.z); o.w = f32_bf16(v.w);
    ((ushort4*)d)[i] = o;
}

// ---------------------------------------------------------------------------
// out[M,N] = A[M,K] * W[N,K]^T   (bf16 in, fp32 accumulate)
// Block tile 128x128, 8 waves (2x4), each wave 64x32 -> 4x2 WMMA subtiles.
// Double-buffered LDS staging: tile k+32 is copied (async) while tile k is
// consumed; s_wait_asynccnt 4 retires only the oldest 4 copies (in-order).
// ---------------------------------------------------------------------------
template<bool STORE_BF16, bool ADD_BIAS>
__global__ __launch_bounds__(256)
void gemm_bf16_wmma(const unsigned short* __restrict__ A,
                    const unsigned short* __restrict__ W,
                    float* __restrict__ outF,
                    unsigned short* __restrict__ outB,
                    const float* __restrict__ bias,
                    int M, int N, int K) {
    constexpr int LD   = 40;                     // padded k-stride (elements)
    constexpr int TSZ  = 128 * LD;               // one tile buffer
    __shared__ __align__(16) unsigned short At[2 * TSZ];
    __shared__ __align__(16) unsigned short Bt[2 * TSZ];

    const int tid  = threadIdx.x;
    const int lane = tid & 31;
    const int w    = tid >> 5;
    const int wr   = w >> 2;                     // 0..1
    const int wc   = w & 3;                      // 0..3
    const int half = lane >> 4;
    const int r16  = lane & 15;
    const int m0   = blockIdx.x * 128;
    const int n0   = blockIdx.y * 128;
    const int wm   = wr * 64;
    const int wn   = wc * 32;

    v8f acc[4][2];
#pragma unroll
    for (int i = 0; i < 4; i++)
#pragma unroll
        for (int j = 0; j < 2; j++)
#pragma unroll
            for (int e = 0; e < 8; e++) acc[i][j][e] = 0.0f;

    const int lrow = tid >> 1;                   // 0..127
    const int lk   = (tid & 1) * 16;             // 0 or 16

    auto stage = [&](int k0, int buf) {
        const unsigned short* ga = A + (size_t)(m0 + lrow) * K + k0 + lk;
        const unsigned short* gw = W + (size_t)(n0 + lrow) * K + k0 + lk;
        unsigned short* la = At + buf * TSZ + lrow * LD + lk;
        unsigned short* lb = Bt + buf * TSZ + lrow * LD + lk;
#ifdef USE_ASYNC_LDS
        ASYNC_CP16(ga,     la);
        ASYNC_CP16(ga + 8, la + 8);
        ASYNC_CP16(gw,     lb);
        ASYNC_CP16(gw + 8, lb + 8);
#else
        ((uint4*)la)[0] = ((const uint4*)ga)[0];
        ((uint4*)la)[1] = ((const uint4*)ga)[1];
        ((uint4*)lb)[0] = ((const uint4*)gw)[0];
        ((uint4*)lb)[1] = ((const uint4*)gw)[1];
#endif
    };

    stage(0, 0);
    int buf = 0;
    for (int k0 = 0; k0 < K; k0 += 32) {
        const bool more = (k0 + 32) < K;
        if (more) stage(k0 + 32, buf ^ 1);       // prefetch next tile
#ifdef USE_ASYNC_LDS
        if (more) ASYNC_WAIT(4);                 // oldest 4 (= tile k0) done
        else      ASYNC_WAIT(0);
#endif
        __syncthreads();                         // tile k0 visible to all waves

        const unsigned short* Ab = At + buf * TSZ;
        const unsigned short* Bb = Bt + buf * TSZ;
        Frag16 bfr[2];
#pragma unroll
        for (int ni = 0; ni < 2; ni++) {
            // B 32x16 bf16 layout: lane n=r16, k = half*16 + i  (32B contiguous)
            const unsigned short* rp = Bb + (wn + ni * 16 + r16) * LD + half * 16;
            bfr[ni].u4[0] = ((const uint4*)rp)[0];
            bfr[ni].u4[1] = ((const uint4*)rp)[1];
        }
#pragma unroll
        for (int mi = 0; mi < 4; mi++) {
            // A 16x32 bf16 layout: lane row=r16; k = half*16 + i  (32B contiguous)
            Frag16 af;
            const unsigned short* rp = Ab + (wm + mi * 16 + r16) * LD;
            af.u4[0] = *(const uint4*)(rp + half * 16);
            af.u4[1] = *(const uint4*)(rp + half * 16 + 8);
#pragma unroll
            for (int ni = 0; ni < 2; ni++) {
                acc[mi][ni] = __builtin_amdgcn_wmma_f32_16x16x32_bf16(
                    false, af.bf, false, bfr[ni].bf, (short)0, acc[mi][ni],
                    false, false);
            }
        }
        __syncthreads();                         // all reads of buf done
        buf ^= 1;
    }

    // D layout: m = v + 8*half, n = r16
#pragma unroll
    for (int mi = 0; mi < 4; mi++)
#pragma unroll
        for (int ni = 0; ni < 2; ni++) {
            const int col = n0 + wn + ni * 16 + r16;
#pragma unroll
            for (int v = 0; v < 8; v++) {
                const int row = m0 + wm + mi * 16 + half * 8 + v;
                float fv = acc[mi][ni][v];
                if (ADD_BIAS) fv += bias[col];
                if (STORE_BF16) outB[(size_t)row * N + col] = f32_bf16(fv);
                else            outF[(size_t)row * N + col] = fv;
            }
        }
}

// ---------------------------------------------------------------------------
// Flash attention, S^T formulation.
//   qkv : (B*T) x 3072 bf16 rows = [Q | K | V], each 1024 wide (16 heads x 64).
// Block = 8 waves; wave w owns queries [q0, q0+16), q0 = bx*128 + w*16.
//
// S^T = K·Q^T via WMMA: its D-layout (m = key = v+8*half, n = query = lane&15)
// is exactly the A-operand layout for P in O += P·V (transpose is free).
// Softmax stats are one-query-per-lane: VALU reduce + one v_permlanex16.
// V blocks are double-buffered per wave with async copies; the next block's
// copy overlaps the current block's S^T WMMAs + softmax.
// ---------------------------------------------------------------------------
__global__ __launch_bounds__(256)
void flash_attn_wmma(const unsigned short* __restrict__ qkv,
                     unsigned short* __restrict__ attn_out) {
    constexpr int T    = 2048;
    constexpr int SQ   = 3072;   // qkv row stride (elements)
    constexpr int VST  = 72;     // V LDS row stride (64 + pad)
    constexpr int VBUF = 32 * VST;
    __shared__ __align__(16) unsigned short Vlds[8 * 2 * VBUF];

    const int tid  = threadIdx.x;
    const int lane = tid & 31;
    const int w    = tid >> 5;
    const int half = lane >> 4;
    const int r16  = lane & 15;
    const int bh   = blockIdx.y;
    const int b    = bh >> 4;           // H = 16
    const int h    = bh & 15;
    const int q0   = blockIdx.x * 128 + w * 16;

    unsigned short* Vw = Vlds + w * 2 * VBUF;

    auto stageV = [&](int j0, int buf) {
        const unsigned short* vp =
            qkv + (size_t)(b * T + j0 + lane) * SQ + 2048 + h * 64;
        unsigned short* vl = Vw + buf * VBUF + lane * VST;
#ifdef USE_ASYNC_LDS
#pragma unroll
        for (int u = 0; u < 4; u++) ASYNC_CP16(vp + 8 * u, vl + 8 * u);
#else
#pragma unroll
        for (int u = 0; u < 4; u++)
            ((uint4*)vl)[u] = ((const uint4*)vp)[u];
#endif
    };

    // Q^T as B-fragments (resident): chunk c covers d = 32c..32c+31.
    // B layout: lane n = r16 (query), k = half*16 + i  -> 32B contiguous.
    Frag16 bQ[2];
    {
        const unsigned short* qp = qkv + (size_t)(b * T + q0 + r16) * SQ + h * 64;
#pragma unroll
        for (int c = 0; c < 2; c++) {
            bQ[c].u4[0] = *(const uint4*)(qp + 32 * c + half * 16);
            bQ[c].u4[1] = *(const uint4*)(qp + 32 * c + half * 16 + 8);
        }
    }

    const int   qidx  = q0 + r16;       // this lane's query (replicated per half)
    const float scale = 0.125f;         // 1/sqrt(64)
    float m_q = -__builtin_huge_valf();
    float l_q = 0.0f;
    v8f acc[4];
#pragma unroll
    for (int dt = 0; dt < 4; dt++)
#pragma unroll
        for (int e = 0; e < 8; e++) acc[dt][e] = 0.0f;

    stageV(0, 0);                       // prologue: first V block in flight
    int vbuf = 0;

    for (int j0 = 0; j0 < q0 + 16; j0 += 32) {
        const bool more = (j0 + 32) < (q0 + 16);
        if (more) stageV(j0 + 32, vbuf ^ 1);     // prefetch next V block

        // ---- S^T = K · Q^T : all 4 K fragments first, then 4 WMMAs --------
        // A layout: lane row = key (j0 + nt*16 + r16);
        // k(i) = 32c + half*16 + i (32B contiguous).
        Frag16 aK[2][2];
#pragma unroll
        for (int nt = 0; nt < 2; nt++) {
            const unsigned short* kp =
                qkv + (size_t)(b * T + j0 + nt * 16 + r16) * SQ + 1024 + h * 64;
#pragma unroll
            for (int c = 0; c < 2; c++) {
                aK[nt][c].u4[0] = *(const uint4*)(kp + 32 * c + half * 16);
                aK[nt][c].u4[1] = *(const uint4*)(kp + 32 * c + half * 16 + 8);
            }
        }
        v8f st[2];
#pragma unroll
        for (int nt = 0; nt < 2; nt++) {
#pragma unroll
            for (int e = 0; e < 8; e++) st[nt][e] = 0.0f;
#pragma unroll
            for (int c = 0; c < 2; c++)
                st[nt] = __builtin_amdgcn_wmma_f32_16x16x32_bf16(
                    false, aK[nt][c].bf, false, bQ[c].bf, (short)0, st[nt],
                    false, false);
        }

        // ---- causal mask + scale; per-lane row max over this block --------
        // element (nt, v): key = j0 + nt*16 + 8*half + v, query = qidx
        float sv[2][8];
        float rm = -__builtin_huge_valf();
#pragma unroll
        for (int nt = 0; nt < 2; nt++)
#pragma unroll
            for (int v = 0; v < 8; v++) {
                const int key = j0 + nt * 16 + 8 * half + v;
                const float x = (key <= qidx) ? st[nt][v] * scale
                                              : -__builtin_huge_valf();
                sv[nt][v] = x;
                rm = fmaxf(rm, x);
            }
        rm = fmaxf(rm, xhalf(rm));                  // combine the two halves

        const float mn = fmaxf(m_q, rm);
        const float a  = __expf(m_q - mn);

        // ---- P = exp(S - m) straight into the A-fragment (no LDS) ---------
        Frag16 aP;
        float rs = 0.0f;
#pragma unroll
        for (int nt = 0; nt < 2; nt++)
#pragma unroll
            for (int v = 0; v < 8; v++) {
                const float e = __expf(sv[nt][v] - mn);
                rs += e;
                aP.us[nt * 8 + v] = f32_bf16(e);
            }
        rs += xhalf(rs);
        l_q = l_q * a + rs;
        m_q = mn;

        // ---- rescale O (row-indexed alpha: row v of acc = query 8*half+v) -
        float arow[8];
#pragma unroll
        for (int v = 0; v < 8; v++) arow[v] = __shfl(a, 8 * half + v, 32);
#pragma unroll
        for (int dt = 0; dt < 4; dt++)
#pragma unroll
            for (int v = 0; v < 8; v++) acc[dt][v] *= arow[v];

#ifdef USE_ASYNC_LDS
        // Oldest 4 async ops are this block's V copy (in-order completion);
        // leave the prefetch (next 4) in flight.
        if (more) ASYNC_WAIT(4);
        else      ASYNC_WAIT(0);
#endif
        // ---- O += P · V (4 d-subtiles, contraction over 32 keys) ----------
        const unsigned short* Vb = Vw + vbuf * VBUF;
#pragma unroll
        for (int dt = 0; dt < 4; dt++) {
            Frag16 bV;                  // B layout: b[i] = V[half*8+i+(i>=8?8:0)][dt*16+r16]
#pragma unroll
            for (int i = 0; i < 16; i++)
                bV.us[i] = Vb[(half * 8 + i + (i >= 8 ? 8 : 0)) * VST + dt * 16 + r16];
            acc[dt] = __builtin_amdgcn_wmma_f32_16x16x32_bf16(
                false, aP.bf, false, bV.bf, (short)0, acc[dt], false, false);
        }
        vbuf ^= 1;
    }

    // ---- normalize (broadcast 1/l to row-indexed layout) and store --------
    float lrow[8];
#pragma unroll
    for (int v = 0; v < 8; v++) lrow[v] = __shfl(l_q, 8 * half + v, 32);
#pragma unroll
    for (int dt = 0; dt < 4; dt++)
#pragma unroll
        for (int v = 0; v < 8; v++) {
            const float o   = acc[dt][v] / lrow[v];
            const size_t rr = (size_t)(b * T + q0 + 8 * half + v);
            attn_out[rr * 1024 + h * 64 + dt * 16 + r16] = f32_bf16(o);
        }
}

// ---------------------------------------------------------------------------
// Launcher
// ---------------------------------------------------------------------------
extern "C" void kernel_launch(void* const* d_in, const int* in_sizes, int n_in,
                              void* d_out, int out_size, void* d_ws, size_t ws_size,
                              hipStream_t stream) {
    (void)in_sizes; (void)n_in; (void)out_size; (void)ws_size;
    const float* x      = (const float*)d_in[0];
    const float* w_qkv  = (const float*)d_in[1];
    const float* w_proj = (const float*)d_in[2];
    const float* b_proj = (const float*)d_in[3];
    float* out = (float*)d_out;

    const int Bq = 4, T = 2048, C = 1024, H = 16;
    const int M = Bq * T;                     // 8192

    char* p = (char*)d_ws;
    unsigned short* x_bf    = (unsigned short*)p; p += (size_t)M * C * 2;
    unsigned short* wqkv_bf = (unsigned short*)p; p += (size_t)3 * C * C * 2;
    unsigned short* wproj_bf= (unsigned short*)p; p += (size_t)C * C * 2;
    unsigned short* qkv_bf  = (unsigned short*)p; p += (size_t)M * 3 * C * 2;
    unsigned short* attn_bf = (unsigned short*)p;

    auto cast = [&](const float* s, unsigned short* d, size_t n) {
        int n4 = (int)(n / 4);
        cast_f32_to_bf16_v4<<<(n4 + 255) / 256, 256, 0, stream>>>(s, d, n4);
    };
    cast(x,      x_bf,     (size_t)M * C);
    cast(w_qkv,  wqkv_bf,  (size_t)3 * C * C);
    cast(w_proj, wproj_bf, (size_t)C * C);

    // QKV projection: [8192,1024] x [3072,1024]^T -> bf16 [8192,3072]
    gemm_bf16_wmma<true, false><<<dim3(M / 128, (3 * C) / 128), 256, 0, stream>>>(
        x_bf, wqkv_bf, nullptr, qkv_bf, nullptr, M, 3 * C, C);

    // Flash attention -> bf16 [8192,1024]
    flash_attn_wmma<<<dim3(T / 128, Bq * H), 256, 0, stream>>>(qkv_bf, attn_bf);

    // Output projection: [8192,1024] x [1024,1024]^T + bias -> fp32 d_out
    gemm_bf16_wmma<false, true><<<dim3(M / 128, C / 128), 256, 0, stream>>>(
        attn_bf, wproj_bf, out, nullptr, b_proj, M, C, C);
}
